// PhasorLayerLinear_3195455668497
// MI455X (gfx1250) — compile-verified, matrix operand order checked by reading, not run on hardware
//
#include <hip/hip_runtime.h>
#include <hip/hip_bf16.h>
#include <math.h>

// ---------------------------------------------------------------------------
// PhasorLayerLinear on MI455X (gfx1250): bf16 WMMA + TDM async weight staging.
// B=65536, DIM=1024, N_PLANES=16.
//
// Pipeline:
//   K0  cvt:      weights fp32 -> bf16 (L2-resident, 4 MB total)
//   K1  phase:    WMMA key/query projections -> per-row coef; writes bf16 x
//   K2  valgemm:  mid = (xb @ value_W.T + value_b) * coef
//                 [8 waves/slab; B tiles via tensor_load_to_lds, dbl-buffered]
//   K3  ln:       LayerNorm(mid) -> bf16 normed
//   K4  outgemm:  out = x + normed @ out_W.T + out_b        [same TDM scheme]
// ---------------------------------------------------------------------------

#define BDIM   65536
#define DIM    1024
#define NPL    16
#define PI_F   3.14159265358979323846f

// LDS tile: 128 rows x 32 bf16 cols = 64B/row, padded by TDM to 80B/row.
#define TILE_ROWS       128
#define TILE_K          32
#define ROW_ELEMS_PAD   40              // 80 bytes / 2
#define BUF_ELEMS       (TILE_ROWS * ROW_ELEMS_PAD)   // 5120 bf16 = 10240 B

typedef __bf16 bf16_t;
typedef bf16_t v16bf __attribute__((ext_vector_type(16)));
typedef bf16_t v8bf  __attribute__((ext_vector_type(8)));
typedef bf16_t v4bf  __attribute__((ext_vector_type(4)));
typedef float  v8f   __attribute__((ext_vector_type(8)));
typedef unsigned int u32x4 __attribute__((ext_vector_type(4)));
typedef int          i32x4 __attribute__((ext_vector_type(4)));
typedef int          i32x8 __attribute__((ext_vector_type(8)));

union V16U  { v16bf v; v8bf h[2]; };
union V16U4 { v16bf v; v4bf q[4]; };

// fp32 -> bf16 round-to-nearest-even
static __device__ __forceinline__ bf16_t f2bf(float f) {
    unsigned u = __builtin_bit_cast(unsigned, f);
    u += 0x7FFFu + ((u >> 16) & 1u);
    unsigned short hs = (unsigned short)(u >> 16);
    return __builtin_bit_cast(bf16_t, hs);
}

static __device__ __forceinline__ v8bf cvt8(const float* __restrict__ p) {
    v8bf r;
#pragma unroll
    for (int i = 0; i < 8; ++i) r[i] = f2bf(p[i]);
    return r;
}

static __device__ __forceinline__ v8bf ld8(const bf16_t* __restrict__ p) {
    return *reinterpret_cast<const v8bf*>(p);
}
static __device__ __forceinline__ v4bf ld4(const bf16_t* p) {
    return *reinterpret_cast<const v4bf*>(p);
}
static __device__ __forceinline__ void st8(bf16_t* __restrict__ p, v8bf v) {
    *reinterpret_cast<v8bf*>(p) = v;
}

// ---------------------------------------------------------------------------
// TDM: async-load one 128x32 bf16 tile (row stride DIM) from global into LDS,
// padding each 64B row with 16B so LDS row stride is 80B (bank skew).
// D# layout per CDNA5 ISA ch.8 (group0: control/lds/global addr; group1: dims)
// This toolchain's builtin arity: (g0, g1, g2, g3, g4, cpol).
// ---------------------------------------------------------------------------
static __device__ __forceinline__ void tdm_issue_b(const bf16_t* gsrc, unsigned ldsAddr) {
    unsigned long long ga = (unsigned long long)(uintptr_t)gsrc;
    u32x4 g0;
    g0[0] = 1u;                                   // count=1, user descriptor
    g0[1] = ldsAddr;                              // lds_addr
    g0[2] = (unsigned)ga;                         // global_addr[31:0]
    g0[3] = ((unsigned)(ga >> 32) & 0x01FFFFFFu)  // global_addr[56:32]
            | 0x80000000u;                        // type=2 ("image")
    i32x8 g1;
    g1[0] = (int)((1u << 16)      // data_size = 2 bytes
                | (1u << 20)      // pad_enable
                | (3u << 22)      // pad_interval: 16 dwords (64B)
                | (3u << 25));    // pad_amount:   4 dwords (16B)
    g1[1] = (int)(1024u << 16);   // tensor_dim0 = 1024 (bits[79:48] lo16)
    g1[2] = (int)(1024u << 16);   // tensor_dim0 hi=0 | tensor_dim1 lo16 = 1024
    g1[3] = (int)(32u << 16);     // tensor_dim1 hi=0 | tile_dim0 = 32
    g1[4] = (int)TILE_ROWS;       // tile_dim1 = 128, tile_dim2 = 0
    g1[5] = (int)DIM;             // tensor_dim0_stride lo32 = 1024
    g1[6] = 0;                    // stride hi | tensor_dim1_stride lo
    g1[7] = 0;
    i32x4 z4 = {0, 0, 0, 0};              // groups 2/3 unused (2D tensor)
    i32x8 z8 = {0, 0, 0, 0, 0, 0, 0, 0};
    __builtin_amdgcn_tensor_load_to_lds(g0, g1, z4, z4, z8, 0);
}

// One K-step of the 16x128 WMMA tile: A from global bf16, B from LDS tile.
static __device__ __forceinline__ void gemm_step(const bf16_t* __restrict__ arow,
                                                 const bf16_t* wbuf, int k0,
                                                 int l15, int kSel, v8f acc[8]) {
    const int kA1 = k0 + kSel * 8;
    const int kA2 = k0 + 16 + kSel * 8;
    V16U a; a.h[0] = ld8(arow + kA1); a.h[1] = ld8(arow + kA2);
#pragma unroll
    for (int t = 0; t < 8; ++t) {
        const bf16_t* rp = wbuf + (t * 16 + l15) * ROW_ELEMS_PAD;
        V16U4 b;
        b.q[0] = ld4(rp + 8 * kSel);
        b.q[1] = ld4(rp + 8 * kSel + 4);
        b.q[2] = ld4(rp + 16 + 8 * kSel);
        b.q[3] = ld4(rp + 16 + 8 * kSel + 4);
        acc[t] = __builtin_amdgcn_wmma_f32_16x16x32_bf16(false, a.v, false, b.v,
                                                         (short)0, acc[t], false, false);
    }
}

// ---------------------------------------------------------------------------
// K0: fp32 -> bf16 weight conversion
// ---------------------------------------------------------------------------
__global__ void cvt_kernel(const float* __restrict__ in, bf16_t* __restrict__ out, int n) {
    int i = blockIdx.x * blockDim.x + threadIdx.x;
    if (i < n) out[i] = f2bf(in[i]);
}

// ---------------------------------------------------------------------------
// K1: phase projections + alignment + gain -> per-row coefficient; also
// writes the bf16 copy of x consumed by the GEMMs. One wave per 16 rows.
// ---------------------------------------------------------------------------
__global__ void __launch_bounds__(32)
phase_kernel(const float* __restrict__ x,
             const bf16_t* __restrict__ kWb, const bf16_t* __restrict__ qWb,
             const float* __restrict__ kb,   const float* __restrict__ qb,
             float* __restrict__ coef,       bf16_t* __restrict__ xb) {
    const int lane = threadIdx.x;
    const int l15  = lane & 15;
    const int kSel = lane >> 4;
    const int rowBase = blockIdx.x * 16;

    const float* xrow = x + (size_t)(rowBase + l15) * DIM;
    bf16_t*      brow = xb + (size_t)(rowBase + l15) * DIM;
    const bf16_t* krow = kWb + (size_t)l15 * DIM;
    const bf16_t* qrow = qWb + (size_t)l15 * DIM;

    v8f accK = {}, accQ = {};

    for (int k0 = 0; k0 < DIM; k0 += 32) {
        const int kA1 = k0 + kSel * 8;
        const int kA2 = k0 + 16 + kSel * 8;
        if (k0 + 32 < DIM) __builtin_prefetch(xrow + k0 + 32, 0, 1);

        V16U a;  a.h[0] = cvt8(xrow + kA1); a.h[1] = cvt8(xrow + kA2);
        st8(brow + kA1, a.h[0]);   // lanes (l15,l15+16) jointly cover all 32 K
        st8(brow + kA2, a.h[1]);

        V16U bk; bk.h[0] = ld8(krow + kA1); bk.h[1] = ld8(krow + kA2);
        V16U bq; bq.h[0] = ld8(qrow + kA1); bq.h[1] = ld8(qrow + kA2);

        accK = __builtin_amdgcn_wmma_f32_16x16x32_bf16(false, a.v, false, bk.v,
                                                       (short)0, accK, false, false);
        accQ = __builtin_amdgcn_wmma_f32_16x16x32_bf16(false, a.v, false, bq.v,
                                                       (short)0, accQ, false, false);
    }

    const float kbv = kb[l15];
    const float qbv = qb[l15];

#pragma unroll
    for (int r = 0; r < 8; ++r) {
        float kp = PI_F * tanhf(accK[r] + kbv);
        float qp = PI_F * tanhf(accQ[r] + qbv);
        float al = cosf(kp - qp);
        al += __shfl_xor(al, 1, 32);
        al += __shfl_xor(al, 2, 32);
        al += __shfl_xor(al, 4, 32);
        al += __shfl_xor(al, 8, 32);
        if (l15 == 0) {
            float s = al;
            float z = s / (float)NPL + 0.5f;
            float gain = (z > 20.0f) ? z : log1pf(expf(z));
            coef[rowBase + r + 8 * kSel] = s * gain / (float)NPL;
        }
    }
}

// ---------------------------------------------------------------------------
// K2: mid = (xb @ value_W.T + value_b) * coef[row]
// 8 waves per 16-row slab; wave w owns cols [w*128, w*128+128) and
// double-buffers its weight tiles in LDS via the Tensor Data Mover.
// ---------------------------------------------------------------------------
__global__ void __launch_bounds__(256)
value_gemm_kernel(const bf16_t* __restrict__ xb, const bf16_t* __restrict__ Wb,
                  const float* __restrict__ vbias, const float* __restrict__ coef,
                  float* __restrict__ mid) {
    __shared__ __align__(16) bf16_t smem[2 * 8 * BUF_ELEMS];   // 160 KB

    const int lane  = threadIdx.x & 31;
    const int waveu = __builtin_amdgcn_readfirstlane((int)(threadIdx.x >> 5));
    const int l15   = lane & 15;
    const int kSel  = lane >> 4;
    const int rowBase = blockIdx.x * 16;
    const int nBase   = waveu * 128;

    bf16_t* buf0 = &smem[(waveu * 2 + 0) * BUF_ELEMS];
    bf16_t* buf1 = &smem[(waveu * 2 + 1) * BUF_ELEMS];
    const unsigned lds0 = (unsigned)(uintptr_t)buf0;
    const unsigned lds1 = (unsigned)(uintptr_t)buf1;

    const bf16_t* wslab = Wb + (size_t)nBase * DIM;
    const bf16_t* arow  = xb + (size_t)(rowBase + l15) * DIM;

    v8f acc[8];
#pragma unroll
    for (int t = 0; t < 8; ++t) acc[t] = (v8f){};

    tdm_issue_b(wslab, lds0);                       // step 0 -> buf0
    for (int i = 0; i < 31; ++i) {
        const int k0 = i * 32;
        tdm_issue_b(wslab + (k0 + 32), (i & 1) ? lds0 : lds1);  // step i+1
        __builtin_amdgcn_s_wait_tensorcnt(1);       // step i tile resident
        __builtin_prefetch(arow + k0 + 32, 0, 1);
        gemm_step(arow, (i & 1) ? buf1 : buf0, k0, l15, kSel, acc);
    }
    __builtin_amdgcn_s_wait_tensorcnt(0);
    gemm_step(arow, buf1, 31 * 32, l15, kSel, acc);

#pragma unroll
    for (int r = 0; r < 8; ++r) {
        const int row = rowBase + r + 8 * kSel;
        const float cf = coef[row];
        float* orow = mid + (size_t)row * DIM;
#pragma unroll
        for (int t = 0; t < 8; ++t) {
            const int col = nBase + t * 16 + l15;
            orow[col] = (acc[t][r] + vbias[col]) * cf;
        }
    }
}

// ---------------------------------------------------------------------------
// K3: LayerNorm over each row of mid -> bf16 normed
// ---------------------------------------------------------------------------
__global__ void __launch_bounds__(256)
ln_kernel(const float* __restrict__ mid, const float* __restrict__ g,
          const float* __restrict__ bta, bf16_t* __restrict__ normed) {
    const int row = blockIdx.x;
    const int t   = threadIdx.x;
    const float* p = mid + (size_t)row * DIM;

    float v[4], s = 0.f, ss = 0.f;
#pragma unroll
    for (int i = 0; i < 4; ++i) {
        float f = p[t + 256 * i];
        v[i] = f; s += f; ss += f * f;
    }
#pragma unroll
    for (int m = 1; m < 32; m <<= 1) {
        s  += __shfl_xor(s,  m, 32);
        ss += __shfl_xor(ss, m, 32);
    }
    __shared__ float sbuf[8], ssbuf[8];
    const int wave = t >> 5;
    if ((t & 31) == 0) { sbuf[wave] = s; ssbuf[wave] = ss; }
    __syncthreads();
    float S = 0.f, SS = 0.f;
#pragma unroll
    for (int w = 0; w < 8; ++w) { S += sbuf[w]; SS += ssbuf[w]; }

    const float mu   = S / (float)DIM;
    const float var  = SS / (float)DIM - mu * mu;
    const float rstd = rsqrtf(var + 1e-5f);

    bf16_t* orow = normed + (size_t)row * DIM;
#pragma unroll
    for (int i = 0; i < 4; ++i) {
        const int col = t + 256 * i;
        orow[col] = f2bf((v[i] - mu) * rstd * g[col] + bta[col]);
    }
}

// ---------------------------------------------------------------------------
// K4: out = x + normed @ out_W.T + out_b   (bf16 WMMA + TDM, fp32 out)
// ---------------------------------------------------------------------------
__global__ void __launch_bounds__(256)
out_gemm_kernel(const bf16_t* __restrict__ normed, const bf16_t* __restrict__ Wb,
                const float* __restrict__ x, const float* __restrict__ obias,
                float* __restrict__ out) {
    __shared__ __align__(16) bf16_t smem[2 * 8 * BUF_ELEMS];   // 160 KB

    const int lane  = threadIdx.x & 31;
    const int waveu = __builtin_amdgcn_readfirstlane((int)(threadIdx.x >> 5));
    const int l15   = lane & 15;
    const int kSel  = lane >> 4;
    const int rowBase = blockIdx.x * 16;
    const int nBase   = waveu * 128;

    bf16_t* buf0 = &smem[(waveu * 2 + 0) * BUF_ELEMS];
    bf16_t* buf1 = &smem[(waveu * 2 + 1) * BUF_ELEMS];
    const unsigned lds0 = (unsigned)(uintptr_t)buf0;
    const unsigned lds1 = (unsigned)(uintptr_t)buf1;

    const bf16_t* wslab = Wb + (size_t)nBase * DIM;
    const bf16_t* arow  = normed + (size_t)(rowBase + l15) * DIM;

    v8f acc[8];
#pragma unroll
    for (int t = 0; t < 8; ++t) acc[t] = (v8f){};

    tdm_issue_b(wslab, lds0);
    for (int i = 0; i < 31; ++i) {
        const int k0 = i * 32;
        tdm_issue_b(wslab + (k0 + 32), (i & 1) ? lds0 : lds1);
        __builtin_amdgcn_s_wait_tensorcnt(1);
        __builtin_prefetch(arow + k0 + 32, 0, 1);
        gemm_step(arow, (i & 1) ? buf1 : buf0, k0, l15, kSel, acc);
    }
    __builtin_amdgcn_s_wait_tensorcnt(0);
    gemm_step(arow, buf1, 31 * 32, l15, kSel, acc);

#pragma unroll
    for (int r = 0; r < 8; ++r) {
        const int row = rowBase + r + 8 * kSel;
        const float* xr = x + (size_t)row * DIM;
        float* orow = out + (size_t)row * DIM;
#pragma unroll
        for (int t = 0; t < 8; ++t) {
            const int col = nBase + t * 16 + l15;
            orow[col] = xr[col] + acc[t][r] + obias[col];
        }
    }
}

// ---------------------------------------------------------------------------
// Host launcher
// ---------------------------------------------------------------------------
static inline size_t align256(size_t o) { return (o + 255u) & ~(size_t)255u; }

extern "C" void kernel_launch(void* const* d_in, const int* in_sizes, int n_in,
                              void* d_out, int out_size, void* d_ws, size_t ws_size,
                              hipStream_t stream) {
    // input order: x, key_W, key_b, query_W, query_b, value_W, value_b,
    //              ln_g, ln_b, out_W, out_b
    const float* x       = (const float*)d_in[0];
    const float* key_W   = (const float*)d_in[1];
    const float* key_b   = (const float*)d_in[2];
    const float* query_W = (const float*)d_in[3];
    const float* query_b = (const float*)d_in[4];
    const float* value_W = (const float*)d_in[5];
    const float* value_b = (const float*)d_in[6];
    const float* ln_g    = (const float*)d_in[7];
    const float* ln_b    = (const float*)d_in[8];
    const float* out_W   = (const float*)d_in[9];
    const float* out_b   = (const float*)d_in[10];
    float* out = (float*)d_out;

    // workspace carve
    char* ws = (char*)d_ws;
    size_t off = 0;
    bf16_t* valWb = (bf16_t*)(ws + off); off = align256(off + (size_t)DIM * DIM * sizeof(bf16_t));
    bf16_t* outWb = (bf16_t*)(ws + off); off = align256(off + (size_t)DIM * DIM * sizeof(bf16_t));
    bf16_t* keyWb = (bf16_t*)(ws + off); off = align256(off + (size_t)NPL * DIM * sizeof(bf16_t));
    bf16_t* qryWb = (bf16_t*)(ws + off); off = align256(off + (size_t)NPL * DIM * sizeof(bf16_t));
    float*  coef  = (float*) (ws + off); off = align256(off + (size_t)BDIM * sizeof(float));
    bf16_t* xb    = (bf16_t*)(ws + off); off = align256(off + (size_t)BDIM * DIM * sizeof(bf16_t));
    float*  mid   = (float*) (ws + off); off = align256(off + (size_t)BDIM * DIM * sizeof(float));
    bf16_t* nrm   = (bf16_t*)(ws + off); off = align256(off + (size_t)BDIM * DIM * sizeof(bf16_t));
    (void)ws_size; (void)n_in; (void)in_sizes; (void)out_size;

    // K0: weight conversions
    {
        const int n1 = DIM * DIM, n2 = NPL * DIM;
        cvt_kernel<<<(n1 + 255) / 256, 256, 0, stream>>>(value_W, valWb, n1);
        cvt_kernel<<<(n1 + 255) / 256, 256, 0, stream>>>(out_W,   outWb, n1);
        cvt_kernel<<<(n2 + 255) / 256, 256, 0, stream>>>(key_W,   keyWb, n2);
        cvt_kernel<<<(n2 + 255) / 256, 256, 0, stream>>>(query_W, qryWb, n2);
    }

    // K1: per-row alignment/gain coefficient + bf16 x copy
    phase_kernel<<<BDIM / 16, 32, 0, stream>>>(x, keyWb, qryWb, key_b, query_b, coef, xb);

    // K2: value GEMM + scale (TDM-staged weights)
    value_gemm_kernel<<<BDIM / 16, 256, 0, stream>>>(xb, valWb, value_b, coef, mid);

    // K3: LayerNorm -> bf16
    ln_kernel<<<BDIM, 256, 0, stream>>>(mid, ln_g, ln_b, nrm);

    // K4: output GEMM + residual + bias (TDM-staged weights)
    out_gemm_kernel<<<BDIM / 16, 256, 0, stream>>>(nrm, outWb, x, out_b, out);
}
